// GraphTransformerLayer_72318659330102
// MI455X (gfx1250) — matrix operand order repaired and checked by reading, not compiled, not run on hardware
//
#include <hip/hip_runtime.h>
#include <hip/hip_bf16.h>

#define N_NODES 6144
#define EDGES   196608
#define HID     256
#define NHEADS  4
#define HDIM    64
#define NH_ELEM (N_NODES * HID)
#define E16     (EDGES * 16)

typedef __attribute__((ext_vector_type(16))) __bf16 v16bf;
typedef __attribute__((ext_vector_type(8)))  float  v8f;

static __device__ __forceinline__ v8f wmma_bf16(v16bf a, v16bf b, v8f c) {
    return __builtin_amdgcn_wmma_f32_16x16x32_bf16(false, a, false, b, (short)0, c, false, false);
}

static __device__ __forceinline__ float gelu_exact(float x) {
    return 0.5f * x * (1.0f + erff(x * 0.7071067811865475f));
}

static __device__ __forceinline__ void atomicMaxFloat(float* addr, float v) {
    if (v >= 0.0f) atomicMax((int*)addr, __float_as_int(v));
    else           atomicMin((unsigned int*)addr, __float_as_uint(v));
}

// Wait for all outstanding async (DMA-to-LDS) ops of this wave.
static __device__ __forceinline__ void wait_async0() {
#if __has_builtin(__builtin_amdgcn_s_wait_asynccnt)
    __builtin_amdgcn_s_wait_asynccnt(0);
#else
    asm volatile("s_wait_asynccnt 0x0" ::: "memory");
#endif
}

// ---------------------------------------------------------------------------
// Generic C[M,Nc] = A[M,K] @ W[Nc,K]^T + bias, optional exact-GELU epilogue.
// bf16 inputs, f32 accumulate via v_wmma_f32_16x16x32_bf16.
// Block = 128 threads = 4 waves; wave computes a 16x64 tile (4 N-subtiles).
// grid = (Nc/256, M/16)
// ---------------------------------------------------------------------------
__global__ __launch_bounds__(128) void gemm_bf16_kernel(
    const __bf16* __restrict__ A, const __bf16* __restrict__ W,
    const float* __restrict__ bias, float* __restrict__ C,
    int M, int Ncols, int K, int act)
{
    const int lane  = threadIdx.x & 31;
    const int wave  = threadIdx.x >> 5;
    const int m0    = blockIdx.y * 16;
    const int n0    = (blockIdx.x * 4 + wave) * 64;
    const int lrow  = lane & 15;
    const int half  = lane >> 4;
    const int koffA = half * 16;   // A: lo lanes K 0..15, hi lanes K 16..31
    const int koffB = half * 16;   // B: lo lanes K 0..15, hi lanes K 16..31

    v8f zero8 = {};
    v8f acc[4] = {zero8, zero8, zero8, zero8};

    const __bf16* arow = A + (size_t)(m0 + lrow) * K;
    for (int kb = 0; kb < K; kb += 32) {
        v16bf a;
#pragma unroll
        for (int i = 0; i < 8; ++i) {
            a[i]     = arow[kb + koffA + i];
            a[i + 8] = arow[kb + koffA + 8 + i];
        }
#pragma unroll
        for (int t = 0; t < 4; ++t) {
            const __bf16* wrow = W + (size_t)(n0 + t * 16 + lrow) * K + kb + koffB;
            v16bf b;
#pragma unroll
            for (int i = 0; i < 16; ++i) b[i] = wrow[i];
            acc[t] = wmma_bf16(a, b, acc[t]);
        }
    }
#pragma unroll
    for (int t = 0; t < 4; ++t) {
        const int n = n0 + t * 16 + lrow;
        const float bv = bias ? bias[n] : 0.0f;
#pragma unroll
        for (int e = 0; e < 8; ++e) {
            const int m = m0 + half * 8 + e;   // C: VGPR e -> row e (+8 hi lanes)
            float v = acc[t][e] + bv;
            if (act == 1) v = gelu_exact(v);
            C[(size_t)m * Ncols + n] = v;
        }
    }
}

// ---------------------------------------------------------------------------
// Fused multi-head attention over N=6144, 4 heads, head dim 64.
// Block = 128 threads (wave w == head w) handles one 16-query tile.
// K tiles (16 keys x 256 cols, bf16, 8KB) are staged into LDS with
// global_load_async_to_lds_b128 (ASYNCcnt DMA), double-buffered by pairs so
// the DMA for chunk c+1 overlaps the WMMAs/softmax of chunk c. All four
// waves share one staged copy instead of re-reading L2.
// Pass 1: streaming max/denominator (32 keys per reduction round).
// Pass 2: alpha, head-mean attn output, ctx accumulation via WMMA.
// grid = N/16 = 384
// ---------------------------------------------------------------------------
__global__ __launch_bounds__(128) void attn_kernel(
    const __bf16* __restrict__ qbf, const __bf16* __restrict__ kbf,
    const __bf16* __restrict__ vT,  float* __restrict__ ctx,
    float* __restrict__ attn_out)
{
    __shared__ __bf16 sK[4][16][HID];          // two pairs of staged K tiles (32 KB)
    __shared__ float  sML[NHEADS][16][2];
    __shared__ float  sAlpha[NHEADS][16][32];
    __shared__ float  sAttn[16][32];

    const int tid   = threadIdx.x;
    const int h     = tid >> 5;        // head == wave
    const int lane  = tid & 31;
    const int lrow  = lane & 15;
    const int half  = lane >> 4;
    const int koffA = half * 8;
    const int koffB = half * 16;
    const int q0    = blockIdx.x * 16;
    const float scale = 0.125f;        // 1/sqrt(64)
    const int NT2 = N_NODES / 32;      // 32-key chunks

    // Stage one 16x256 bf16 K tile (8 KB) into sK[buf] via async DMA:
    // 512 x 16B chunks, 4 per thread.
    auto stage = [&](int kt, int buf) {
#pragma unroll
        for (int i = 0; i < 4; ++i) {
            const int idx = tid + i * 128;            // 0..511
            const int r   = idx >> 5;                 // key row 0..15
            const int cb  = (idx & 31) * 16;          // byte column
            unsigned lds  = (unsigned)(size_t)(&sK[buf][r][0]) + (unsigned)cb;
            const char* g = (const char*)(kbf + (size_t)(kt * 16 + r) * HID) + cb;
            asm volatile("global_load_async_to_lds_b128 %0, %1, off"
                         :: "v"(lds), "v"(g) : "memory");
        }
    };

    // Q fragments (16 queries x 64 dims = two 16x32 A-fragments), resident.
    v16bf qa0, qa1;
    {
        const __bf16* qrow = qbf + (size_t)(q0 + lrow) * HID + h * HDIM;
#pragma unroll
        for (int i = 0; i < 8; ++i) {
            qa0[i]     = qrow[koffB + i];
            qa0[i + 8] = qrow[koffB + 8 + i];
            qa1[i]     = qrow[32 + koffB + i];
            qa1[i + 8] = qrow[32 + koffB + 8 + i];
        }
    }

    // ---- Pass 1: online row max + exp-sum, 32 keys per round ----
    float mr[8], lr[8];
#pragma unroll
    for (int e = 0; e < 8; ++e) { mr[e] = -3.0e38f; lr[e] = 0.0f; }

    stage(0, 0);
    stage(1, 1);
    for (int c2 = 0; c2 < NT2; ++c2) {
        const int pair = (c2 & 1) * 2;               // 0 or 2
        wait_async0();
        __syncthreads();                             // staged pair visible to all waves
        if (c2 + 1 < NT2) {                          // prefetch next pair (overlaps compute)
            stage(2 * c2 + 2, 2 - pair);
            stage(2 * c2 + 3, 3 - pair);
        }

        v8f s0 = {}, s1 = {};
        {
            const __bf16* kr0 = &sK[pair][lrow][h * HDIM + koffB];
            const __bf16* kr1 = &sK[pair + 1][lrow][h * HDIM + koffB];
            v16bf b;
#pragma unroll
            for (int i = 0; i < 16; ++i) b[i] = kr0[i];
            s0 = wmma_bf16(qa0, b, s0);
#pragma unroll
            for (int i = 0; i < 16; ++i) b[i] = kr0[32 + i];
            s0 = wmma_bf16(qa1, b, s0);
#pragma unroll
            for (int i = 0; i < 16; ++i) b[i] = kr1[i];
            s1 = wmma_bf16(qa0, b, s1);
#pragma unroll
            for (int i = 0; i < 16; ++i) b[i] = kr1[32 + i];
            s1 = wmma_bf16(qa1, b, s1);
        }
#pragma unroll
        for (int e = 0; e < 8; ++e) {
            const float v0 = s0[e] * scale;
            const float v1 = s1[e] * scale;
            float tmax = fmaxf(v0, v1);
            for (int o = 1; o < 16; o <<= 1) tmax = fmaxf(tmax, __shfl_xor(tmax, o, 32));
            const float mnew = fmaxf(mr[e], tmax);
            float p = __expf(v0 - mnew) + __expf(v1 - mnew);
            for (int o = 1; o < 16; o <<= 1) p += __shfl_xor(p, o, 32);
            lr[e] = lr[e] * __expf(mr[e] - mnew) + p;
            mr[e] = mnew;
        }
    }
    if (lrow == 0) {
#pragma unroll
        for (int e = 0; e < 8; ++e) {
            sML[h][koffA + e][0] = mr[e];
            sML[h][koffA + e][1] = lr[e];
        }
    }
    __syncthreads();

    // ---- Pass 2: alpha, head-averaged attn output, ctx += alpha @ V ----
    v8f zero8 = {};
    v8f cacc[4] = {zero8, zero8, zero8, zero8};

    stage(0, 0);
    stage(1, 1);
    for (int c2 = 0; c2 < NT2; ++c2) {
        const int pair = (c2 & 1) * 2;
        for (int i = tid; i < 16 * 32; i += 128) ((float*)sAttn)[i] = 0.0f;
        wait_async0();
        __syncthreads();
        if (c2 + 1 < NT2) {
            stage(2 * c2 + 2, 2 - pair);
            stage(2 * c2 + 3, 3 - pair);
        }

#pragma unroll
        for (int t = 0; t < 2; ++t) {
            const __bf16* kr = &sK[pair + t][lrow][h * HDIM + koffB];
            v16bf b0, b1;
#pragma unroll
            for (int i = 0; i < 16; ++i) { b0[i] = kr[i]; b1[i] = kr[32 + i]; }
            v8f s = {};
            s = wmma_bf16(qa0, b0, s);
            s = wmma_bf16(qa1, b1, s);
#pragma unroll
            for (int e = 0; e < 8; ++e) {
                const int m = koffA + e;
                const float alpha = __expf(s[e] * scale - sML[h][m][0]) / sML[h][m][1];
                sAlpha[h][m][t * 16 + lrow] = alpha;
                atomicAdd(&sAttn[m][t * 16 + lrow], 0.25f * alpha);  // mean over 4 heads
            }
        }
        __syncthreads();

        // Write 16x32 head-averaged attention tile.
        for (int i = tid; i < 16 * 32; i += 128) {
            const int r = i >> 5, c = i & 31;
            attn_out[(size_t)(q0 + r) * N_NODES + c2 * 32 + c] = sAttn[r][c];
        }

        // Re-fragment alpha (C-layout -> A-layout) through LDS, as bf16.
        v16bf af;
#pragma unroll
        for (int i = 0; i < 8; ++i) {
            af[i]     = (__bf16)sAlpha[h][lrow][koffB + i];
            af[i + 8] = (__bf16)sAlpha[h][lrow][koffB + 8 + i];
        }

        // ctx(16q x 64d) += alpha(16x32) @ V(32 keys x 64d), via vT[dim][key].
#pragma unroll
        for (int s4 = 0; s4 < 4; ++s4) {
            const __bf16* vrow = vT + (size_t)(h * HDIM + s4 * 16 + lrow) * N_NODES
                                    + c2 * 32 + koffB;
            v16bf bf;
#pragma unroll
            for (int i = 0; i < 16; ++i) bf[i] = vrow[i];
            cacc[s4] = wmma_bf16(af, bf, cacc[s4]);
        }
        __syncthreads();
    }

#pragma unroll
    for (int s4 = 0; s4 < 4; ++s4) {
        const int d = h * HDIM + s4 * 16 + lrow;
#pragma unroll
        for (int e = 0; e < 8; ++e) {
            const int m = koffA + e;
            ctx[(size_t)(q0 + m) * HID + d] = cacc[s4][e];
        }
    }
}

// ---------------------------------------------------------------------------
// Elementwise / reduction helper kernels
// ---------------------------------------------------------------------------
__global__ void cast_f32_bf16_kernel(const float* __restrict__ in,
                                     __bf16* __restrict__ out, int n)
{
    int i = blockIdx.x * blockDim.x + threadIdx.x;
    if (i < n) out[i] = (__bf16)in[i];
}

__global__ void fill_kernel(float* __restrict__ p, float v, int n)
{
    int i = blockIdx.x * blockDim.x + threadIdx.x;
    if (i < n) p[i] = v;
}

__global__ void copy_f32_kernel(const float* __restrict__ in,
                                float* __restrict__ out, int n)
{
    int i = blockIdx.x * blockDim.x + threadIdx.x;
    if (i < n) out[i] = in[i];
}

// qkv [N,768] f32 -> qbf[N,256], kbf[N,256], vT[256][N] (bf16)
__global__ void qkv_cast_kernel(const float* __restrict__ qkv,
                                __bf16* __restrict__ qbf,
                                __bf16* __restrict__ kbf,
                                __bf16* __restrict__ vT, int n)
{
    int i = blockIdx.x * blockDim.x + threadIdx.x;
    if (i >= n) return;
    const int row = i / HID, col = i % HID;
    const float* base = qkv + (size_t)row * (3 * HID);
    qbf[i] = (__bf16)base[col];
    kbf[i] = (__bf16)base[HID + col];
    vT[(size_t)col * N_NODES + row] = (__bf16)base[2 * HID + col];
}

// a_src/a_dst = h @ att vectors, one wave per row
__global__ __launch_bounds__(256) void gat_scores_kernel(
    const float* __restrict__ h, const float* __restrict__ asw,
    const float* __restrict__ adw, float* __restrict__ a_src,
    float* __restrict__ a_dst)
{
    const int wave = threadIdx.x >> 5, lane = threadIdx.x & 31;
    const int row = blockIdx.x * 8 + wave;
    const float* hr = h + (size_t)row * HID;
    float s1 = 0.0f, s2 = 0.0f;
    for (int c = lane; c < HID; c += 32) {
        const float v = hr[c];
        s1 += v * asw[c];
        s2 += v * adw[c];
    }
    for (int o = 16; o > 0; o >>= 1) { s1 += __shfl_xor(s1, o, 32); s2 += __shfl_xor(s2, o, 32); }
    if (lane == 0) { a_src[row] = s1; a_dst[row] = s2; }
}

static __device__ __forceinline__ void edge_sd(const int* ei, int idx, int& s, int& d)
{
    if (idx < EDGES) { s = ei[idx]; d = ei[EDGES + idx]; }
    else             { s = d = idx - EDGES; }
}

__global__ void edge_max_kernel(const int* __restrict__ ei,
                                const float* __restrict__ a_src,
                                const float* __restrict__ a_dst,
                                float* __restrict__ emax)
{
    int idx = blockIdx.x * blockDim.x + threadIdx.x;
    if (idx >= EDGES + N_NODES) return;
    int s, d; edge_sd(ei, idx, s, d);
    float e = a_src[s] + a_dst[d];
    e = (e > 0.0f) ? e : 0.2f * e;      // leaky_relu
    atomicMaxFloat(&emax[d], e);
}

__global__ void edge_exp_kernel(const int* __restrict__ ei,
                                const float* __restrict__ a_src,
                                const float* __restrict__ a_dst,
                                const float* __restrict__ emax,
                                float* __restrict__ eexp,
                                float* __restrict__ denom)
{
    int idx = blockIdx.x * blockDim.x + threadIdx.x;
    if (idx >= EDGES + N_NODES) return;
    int s, d; edge_sd(ei, idx, s, d);
    float e = a_src[s] + a_dst[d];
    e = (e > 0.0f) ? e : 0.2f * e;
    const float ex = __expf(e - emax[d]);
    eexp[idx] = ex;
    atomicAdd(&denom[d], ex);
}

// one wave per edge: xlocal[dst,:] += alpha * h[src,:]
__global__ __launch_bounds__(256) void edge_aggr_kernel(
    const int* __restrict__ ei, const float* __restrict__ eexp,
    const float* __restrict__ denom, const float* __restrict__ h,
    float* __restrict__ xlocal)
{
    const int widx = blockIdx.x * 8 + (threadIdx.x >> 5);
    if (widx >= EDGES + N_NODES) return;
    const int lane = threadIdx.x & 31;
    int s, d; edge_sd(ei, widx, s, d);
    const float alpha = eexp[widx] / denom[d];
    const float* hs = h + (size_t)s * HID;
    float* xd = xlocal + (size_t)d * HID;
    for (int c = lane; c < HID; c += 32) atomicAdd(&xd[c], alpha * hs[c]);
}

// out = LayerNorm(a + b + colbias) * g + beta ; one wave per row of 256
__global__ __launch_bounds__(256) void residual_ln_kernel(
    const float* __restrict__ a, const float* __restrict__ b,
    const float* __restrict__ colbias, const float* __restrict__ g,
    const float* __restrict__ beta, float* __restrict__ out)
{
    const int wave = threadIdx.x >> 5, lane = threadIdx.x & 31;
    const int row = blockIdx.x * 8 + wave;
    const size_t base = (size_t)row * HID;
    float vals[8];
    float sum = 0.0f;
#pragma unroll
    for (int i = 0; i < 8; ++i) {
        const int c = lane * 8 + i;
        float v = a[base + c] + b[base + c];
        if (colbias) v += colbias[c];
        vals[i] = v;
        sum += v;
    }
    for (int o = 16; o > 0; o >>= 1) sum += __shfl_xor(sum, o, 32);
    const float mean = sum * (1.0f / HID);
    float var = 0.0f;
#pragma unroll
    for (int i = 0; i < 8; ++i) { const float dd = vals[i] - mean; var += dd * dd; }
    for (int o = 16; o > 0; o >>= 1) var += __shfl_xor(var, o, 32);
    const float inv = rsqrtf(var * (1.0f / HID) + 1e-5f);
#pragma unroll
    for (int i = 0; i < 8; ++i) {
        const int c = lane * 8 + i;
        out[base + c] = (vals[i] - mean) * inv * g[c] + beta[c];
    }
}

// ---------------------------------------------------------------------------
extern "C" void kernel_launch(void* const* d_in, const int* in_sizes, int n_in,
                              void* d_out, int out_size, void* d_ws, size_t ws_size,
                              hipStream_t stream)
{
    const float* x          = (const float*)d_in[0];
    const int*   edge_index = (const int*)  d_in[1];
    const float* edge_attr  = (const float*)d_in[2];
    const float* gat_W      = (const float*)d_in[3];
    const float* att_src_w  = (const float*)d_in[4];
    const float* att_dst_w  = (const float*)d_in[5];
    const float* gat_bias   = (const float*)d_in[6];
    const float* in_proj_w  = (const float*)d_in[7];
    const float* in_proj_b  = (const float*)d_in[8];
    const float* out_proj_w = (const float*)d_in[9];
    const float* out_proj_b = (const float*)d_in[10];
    const float* ffn_w1     = (const float*)d_in[11];
    const float* ffn_b1     = (const float*)d_in[12];
    const float* ffn_w2     = (const float*)d_in[13];
    const float* ffn_b2     = (const float*)d_in[14];
    const float* ln1_g      = (const float*)d_in[15];
    const float* ln1_b      = (const float*)d_in[16];
    const float* ln2_g      = (const float*)d_in[17];
    const float* ln2_b      = (const float*)d_in[18];

    float* out_x    = (float*)d_out;                       // [N,H]
    float* out_edge = out_x + NH_ELEM;                     // [E,16]
    float* out_attn = out_edge + E16;                      // [N,N]

    // --- workspace bump allocator (aliased across phases) ---
    char* ws = (char*)d_ws;
    size_t off = 0;
    auto carve = [&](size_t bytes) -> void* {
        void* p = ws + off;
        off += (bytes + 255) & ~(size_t)255;
        return p;
    };
    float*  bufA   = (float*)carve((size_t)NH_ELEM * 4);            // h -> ctx
    float*  bufB   = (float*)carve((size_t)NH_ELEM * 4);            // xlocal -> xglobal -> xffn
    float*  x1     = (float*)carve((size_t)NH_ELEM * 4);
    float*  x2     = (float*)carve((size_t)NH_ELEM * 4);
    float*  big    = (float*)carve((size_t)N_NODES * 1024 * 4);     // qkv -> ffn mid
    float*  a_src  = (float*)carve((size_t)N_NODES * 4);
    float*  a_dst  = (float*)carve((size_t)N_NODES * 4);
    float*  emax   = (float*)carve((size_t)N_NODES * 4);
    float*  denom  = (float*)carve((size_t)N_NODES * 4);
    float*  eexp   = (float*)carve((size_t)(EDGES + N_NODES) * 4);
    __bf16* abf    = (__bf16*)carve((size_t)N_NODES * 1024 * 2);    // activation bf16 scratch
    __bf16* wbf    = (__bf16*)carve((size_t)1024 * 256 * 2);        // weight bf16 scratch
    __bf16* qbf    = (__bf16*)carve((size_t)NH_ELEM * 2);
    __bf16* kbf    = (__bf16*)carve((size_t)NH_ELEM * 2);
    __bf16* vT     = (__bf16*)carve((size_t)NH_ELEM * 2);

    const int TPB = 256;
    const int nEdgeTot = EDGES + N_NODES;
    const dim3 blk1(TPB);

    // ---- GAT: h = x @ gat_W^T ----
    fill_kernel<<<(N_NODES + TPB - 1) / TPB, blk1, 0, stream>>>(emax, -3.0e38f, N_NODES);
    fill_kernel<<<(N_NODES + TPB - 1) / TPB, blk1, 0, stream>>>(denom, 0.0f, N_NODES);
    fill_kernel<<<(NH_ELEM + TPB - 1) / TPB, blk1, 0, stream>>>(bufB, 0.0f, NH_ELEM);
    cast_f32_bf16_kernel<<<(NH_ELEM + TPB - 1) / TPB, blk1, 0, stream>>>(x, abf, NH_ELEM);
    cast_f32_bf16_kernel<<<(HID * HID + TPB - 1) / TPB, blk1, 0, stream>>>(gat_W, wbf, HID * HID);
    gemm_bf16_kernel<<<dim3(HID / 256, N_NODES / 16), dim3(128), 0, stream>>>(
        abf, wbf, nullptr, bufA, N_NODES, HID, HID, 0);

    gat_scores_kernel<<<N_NODES / 8, blk1, 0, stream>>>(bufA, att_src_w, att_dst_w, a_src, a_dst);
    edge_max_kernel<<<(nEdgeTot + TPB - 1) / TPB, blk1, 0, stream>>>(edge_index, a_src, a_dst, emax);
    edge_exp_kernel<<<(nEdgeTot + TPB - 1) / TPB, blk1, 0, stream>>>(
        edge_index, a_src, a_dst, emax, eexp, denom);
    edge_aggr_kernel<<<(nEdgeTot + 7) / 8, blk1, 0, stream>>>(edge_index, eexp, denom, bufA, bufB);
    residual_ln_kernel<<<N_NODES / 8, blk1, 0, stream>>>(bufB, x, gat_bias, ln1_g, ln1_b, x1);

    // ---- MHA: qkv projection ----
    cast_f32_bf16_kernel<<<(NH_ELEM + TPB - 1) / TPB, blk1, 0, stream>>>(x1, abf, NH_ELEM);
    cast_f32_bf16_kernel<<<(3 * HID * HID + TPB - 1) / TPB, blk1, 0, stream>>>(
        in_proj_w, wbf, 3 * HID * HID);
    gemm_bf16_kernel<<<dim3(3 * HID / 256, N_NODES / 16), dim3(128), 0, stream>>>(
        abf, wbf, in_proj_b, big, N_NODES, 3 * HID, HID, 0);
    qkv_cast_kernel<<<(NH_ELEM + TPB - 1) / TPB, blk1, 0, stream>>>(big, qbf, kbf, vT, NH_ELEM);

    // ---- fused attention: ctx + head-mean attn weights ----
    attn_kernel<<<N_NODES / 16, dim3(128), 0, stream>>>(qbf, kbf, vT, bufA, out_attn);

    // ---- out projection + LN (reference reuses ln1 here) ----
    cast_f32_bf16_kernel<<<(NH_ELEM + TPB - 1) / TPB, blk1, 0, stream>>>(bufA, abf, NH_ELEM);
    cast_f32_bf16_kernel<<<(HID * HID + TPB - 1) / TPB, blk1, 0, stream>>>(
        out_proj_w, wbf, HID * HID);
    gemm_bf16_kernel<<<dim3(HID / 256, N_NODES / 16), dim3(128), 0, stream>>>(
        abf, wbf, out_proj_b, bufB, N_NODES, HID, HID, 0);
    residual_ln_kernel<<<N_NODES / 8, blk1, 0, stream>>>(bufB, x1, nullptr, ln1_g, ln1_b, x2);

    // ---- FFN ----
    cast_f32_bf16_kernel<<<(NH_ELEM + TPB - 1) / TPB, blk1, 0, stream>>>(x2, abf, NH_ELEM);
    cast_f32_bf16_kernel<<<(4 * HID * HID + TPB - 1) / TPB, blk1, 0, stream>>>(
        ffn_w1, wbf, 4 * HID * HID);
    gemm_bf16_kernel<<<dim3(4 * HID / 256, N_NODES / 16), dim3(128), 0, stream>>>(
        abf, wbf, ffn_b1, big, N_NODES, 4 * HID, HID, 1 /*GELU*/);
    cast_f32_bf16_kernel<<<(N_NODES * 1024 + TPB - 1) / TPB, blk1, 0, stream>>>(
        big, abf, N_NODES * 1024);
    cast_f32_bf16_kernel<<<(4 * HID * HID + TPB - 1) / TPB, blk1, 0, stream>>>(
        ffn_w2, wbf, 4 * HID * HID);
    gemm_bf16_kernel<<<dim3(HID / 256, N_NODES / 16), dim3(128), 0, stream>>>(
        abf, wbf, ffn_b2, bufB, N_NODES, HID, 4 * HID, 0);
    residual_ln_kernel<<<N_NODES / 8, blk1, 0, stream>>>(x2, bufB, nullptr, ln2_g, ln2_b, out_x);

    // ---- passthrough edge_attr ----
    copy_f32_kernel<<<(E16 + TPB - 1) / TPB, blk1, 0, stream>>>(edge_attr, out_edge, E16);
}
